// CrossAttention_26886495273177
// MI455X (gfx1250) — matrix operand order repaired and checked by reading, not compiled, not run on hardware
//
#include <hip/hip_runtime.h>
#include <hip/hip_bf16.h>
#include <math.h>

typedef __attribute__((ext_vector_type(16))) _Float16 v16h;
typedef __attribute__((ext_vector_type(8)))  float    v8f;

#define NEG_BIG (-1e30f)

// Convert two float4's into 8 consecutive f16 slots of a v16h (BASE is a literal).
#define CVT8(dst, BASE, a, b) do {                         \
    (dst)[(BASE)+0] = (_Float16)((a).x);                   \
    (dst)[(BASE)+1] = (_Float16)((a).y);                   \
    (dst)[(BASE)+2] = (_Float16)((a).z);                   \
    (dst)[(BASE)+3] = (_Float16)((a).w);                   \
    (dst)[(BASE)+4] = (_Float16)((b).x);                   \
    (dst)[(BASE)+5] = (_Float16)((b).y);                   \
    (dst)[(BASE)+6] = (_Float16)((b).z);                   \
    (dst)[(BASE)+7] = (_Float16)((b).w);                   \
  } while (0)

// One wave32 per 16-query tile of one (batch, head), 32 keys per iteration.
// S^T = K * Q^T via wmma (A = K tiles, B = Q tile), online softmax per query
// (query index == lane&15 in the C layout), then O^T += V^T * P^T via wmma
// with a FULL K=32 contraction (no zero padding).
__global__ __launch_bounds__(32)
void fa_varlen_wmma(const float* __restrict__ q,
                    const float* __restrict__ kv,
                    const int* __restrict__ cu,
                    float* __restrict__ out)
{
  constexpr int H = 16, HKV = 4, D = 64;
  constexpr int KVSTRIDE = 2 * HKV * D;   // floats per token in kv

  const int lane = threadIdx.x & 31;
  const int half = lane >> 4;             // 0: lanes 0-15, 1: lanes 16-31
  const int l16  = lane & 15;

  const int b    = blockIdx.z;
  const int h    = blockIdx.y;
  const int q0   = blockIdx.x * 16;       // local query tile start

  const int seq0 = cu[b];
  const int len  = cu[b + 1] - seq0;
  if (q0 >= len) return;                  // block-uniform exit (EXEC stays full)

  const int hk = h >> 2;                  // H / HKV = 4

  const float* kbase = kv + (size_t)seq0 * KVSTRIDE + hk * D;       // K rows
  const float* vbase = kbase + HKV * D;                             // V rows

  // ---------------- Q tile -> two B matrices (d-chunks of 32), scaled -------
  // B layout (32x16, f16): lane holds col n = l16, K = 8*half + (0..7), 16 + 8*half + (0..7)
  v16h qb[2];
  {
    const int row  = q0 + l16;
    const int rowc = row < len ? row : (len - 1);
    const float* qr = q + ((size_t)(seq0 + rowc) * H + h) * D;
    #pragma unroll
    for (int c = 0; c < 2; ++c) {
      const float4* p = (const float4*)(qr + 32 * c + 8 * half);
      const float4* p2 = (const float4*)(qr + 32 * c + 16 + 8 * half);
      float4 x0 = p[0], x1 = p[1], x2 = p2[0], x3 = p2[1];
      x0.x *= 0.125f; x0.y *= 0.125f; x0.z *= 0.125f; x0.w *= 0.125f;
      x1.x *= 0.125f; x1.y *= 0.125f; x1.z *= 0.125f; x1.w *= 0.125f;
      x2.x *= 0.125f; x2.y *= 0.125f; x2.z *= 0.125f; x2.w *= 0.125f;
      x3.x *= 0.125f; x3.y *= 0.125f; x3.z *= 0.125f; x3.w *= 0.125f;
      CVT8(qb[c], 0, x0, x1);
      CVT8(qb[c], 8, x2, x3);
    }
  }

  // ---------------- online-softmax state + O^T accumulators ------------------
  float mrow = NEG_BIG;                   // per-query running max (query = l16)
  float lrow = 0.0f;                      // per-query running denom
  v8f o[4];                               // O^T chunks: d = 16t + r + 8*half, n = l16
  #pragma unroll
  for (int t = 0; t < 4; ++t)
    #pragma unroll
    for (int r = 0; r < 8; ++r) o[t][r] = 0.0f;

  const int kend = (len < q0 + 16) ? len : (q0 + 16);   // causal key bound
  const int qi   = q0 + l16;

  for (int kb = 0; kb < kend; kb += 32) {
    // ---- prefetch next iteration's K/V rows (uniform branch) ----------------
    if (kb + 32 < kend) {
      const int pk  = kb + 32 + l16;
      const int pkc = pk < len ? pk : (len - 1);
      const float* pr = kbase + (size_t)pkc * KVSTRIDE;
      __builtin_prefetch(pr, 0, 1);
      __builtin_prefetch(pr + HKV * D, 0, 1);
    }

    // ---- K tiles -> four A matrices (2 key-tiles x 2 d-chunks) --------------
    // A layout (16x32, f16): lane row = key l16;
    //   halves 0..7  : d = 32c + 8*half + (0..7)
    //   halves 8..15 : d = 32c + 16 + 8*half + (0..7)
    v16h ka0[2], ka1[2];
    {
      const int key0  = kb + l16;
      const int key0c = key0 < len ? key0 : (len - 1);
      const int key1  = kb + 16 + l16;
      const int key1c = key1 < len ? key1 : (len - 1);
      const float* kr0 = kbase + (size_t)key0c * KVSTRIDE;
      const float* kr1 = kbase + (size_t)key1c * KVSTRIDE;
      #pragma unroll
      for (int c = 0; c < 2; ++c) {
        const float4* p0 = (const float4*)(kr0 + 32 * c + 8 * half);
        const float4* p1 = (const float4*)(kr0 + 32 * c + 16 + 8 * half);
        float4 a0 = p0[0], a1 = p0[1];
        float4 b0 = p1[0], b1 = p1[1];
        CVT8(ka0[c], 0, a0, a1);
        CVT8(ka0[c], 8, b0, b1);
      }
      #pragma unroll
      for (int c = 0; c < 2; ++c) {
        const float4* p0 = (const float4*)(kr1 + 32 * c + 8 * half);
        const float4* p1 = (const float4*)(kr1 + 32 * c + 16 + 8 * half);
        float4 a0 = p0[0], a1 = p0[1];
        float4 b0 = p1[0], b1 = p1[1];
        CVT8(ka1[c], 0, a0, a1);
        CVT8(ka1[c], 8, b0, b1);
      }
    }

    // ---- S^T (32 keys x 16 queries) = K * Q^T, two 16x16 C tiles ------------
    v8f s0 = {0.f, 0.f, 0.f, 0.f, 0.f, 0.f, 0.f, 0.f};
    v8f s1 = {0.f, 0.f, 0.f, 0.f, 0.f, 0.f, 0.f, 0.f};
    s0 = __builtin_amdgcn_wmma_f32_16x16x32_f16(false, ka0[0], false, qb[0],
                                                (short)0, s0, false, false);
    s1 = __builtin_amdgcn_wmma_f32_16x16x32_f16(false, ka1[0], false, qb[0],
                                                (short)0, s1, false, false);
    s0 = __builtin_amdgcn_wmma_f32_16x16x32_f16(false, ka0[1], false, qb[1],
                                                (short)0, s0, false, false);
    s1 = __builtin_amdgcn_wmma_f32_16x16x32_f16(false, ka1[1], false, qb[1],
                                                (short)0, s1, false, false);

    // C layout: s0[r] = S[query = l16][key = kb + r + 8*half]
    //           s1[r] = S[query = l16][key = kb + 16 + r + 8*half]
    // ---- causal / length mask (uniform branch; per-element select) ----------
    if ((kb + 31 > q0) || (kb + 32 > len)) {
      #pragma unroll
      for (int r = 0; r < 8; ++r) {
        const int key0 = kb + r + 8 * half;
        const int key1 = key0 + 16;
        if (key0 > qi || key0 >= len) s0[r] = NEG_BIG;
        if (key1 > qi || key1 >= len) s1[r] = NEG_BIG;
      }
    }

    // ---- online softmax over all 32 keys ------------------------------------
    float tmax = fmaxf(s0[0], s1[0]);
    #pragma unroll
    for (int r = 1; r < 8; ++r) tmax = fmaxf(tmax, fmaxf(s0[r], s1[r]));
    tmax = fmaxf(tmax, __shfl_xor(tmax, 16, 32));
    const float mnew = fmaxf(mrow, tmax);
    const float corr = __expf(mrow - mnew);

    float pf0[8], pf1[8];
    float psum = 0.0f;
    #pragma unroll
    for (int r = 0; r < 8; ++r) {
      pf0[r] = __expf(s0[r] - mnew);
      pf1[r] = __expf(s1[r] - mnew);
      psum += pf0[r] + pf1[r];
    }
    psum += __shfl_xor(psum, 16, 32);
    lrow = lrow * corr + psum;
    mrow = mnew;

    #pragma unroll
    for (int t = 0; t < 4; ++t)
      #pragma unroll
      for (int r = 0; r < 8; ++r) o[t][r] *= corr;

    // ---- build P^T as a FULL 32x16 B matrix ---------------------------------
    // B layout: lanes 0-15 hold keys kb+0..15 (col = l16),
    //           lanes 16-31 hold keys kb+16..31 (col = l16).
    v16h pb;
    #pragma unroll
    for (int r = 0; r < 8; ++r) {
      const float oth0 = __shfl_xor(pf0[r], 16, 32);  // keys kb+8..15 (to low half)
      const float oth1 = __shfl_xor(pf1[r], 16, 32);  // keys kb+16..23 (to high half)
      pb[r]     = (_Float16)(half ? oth1 : pf0[r]);
      pb[r + 8] = (_Float16)(half ? pf1[r] : oth0);
    }

    // ---- O^T += V^T * P^T over 4 d-chunks of 16, full K=32 ------------------
    // A = V^T chunk (16 d-rows x 32 keys): lane row = d-local = l16;
//   halves 0..7  : key = kb + 16*half + j
    //   halves 8..15 : key = kb + 16*half + 8 + j
    #pragma unroll
    for (int t = 0; t < 4; ++t) {
      v16h va;
      #pragma unroll
      for (int j = 0; j < 8; ++j) {
        const int keyA  = kb + 16 * half + j;
        const int keyAc = keyA < len ? keyA : (len - 1);
        const int keyB  = kb + 16 * half + 8 + j;
        const int keyBc = keyB < len ? keyB : (len - 1);
        va[j]     = (_Float16)vbase[(size_t)keyAc * KVSTRIDE + 16 * t + l16];
        va[j + 8] = (_Float16)vbase[(size_t)keyBc * KVSTRIDE + 16 * t + l16];
      }
      o[t] = __builtin_amdgcn_wmma_f32_16x16x32_f16(false, va, false, pb,
                                                    (short)0, o[t], false, false);
    }
  }

  // ---------------- epilogue: O = O^T / l ------------------------------------
  if (qi < len) {
    const float inv = 1.0f / lrow;
    float* op = out + ((size_t)(seq0 + qi) * H + h) * D;
    #pragma unroll
    for (int t = 0; t < 4; ++t)
      #pragma unroll
      for (int r = 0; r < 8; ++r)
        op[16 * t + 8 * half + r] = o[t][r] * inv;   // contiguous 8-float runs
  }
}

extern "C" void kernel_launch(void* const* d_in, const int* in_sizes, int n_in,
                              void* d_out, int out_size, void* d_ws, size_t ws_size,
                              hipStream_t stream) {
  (void)n_in; (void)out_size; (void)d_ws; (void)ws_size;
  const float* q_p  = (const float*)d_in[0];
  const float* kv_p = (const float*)d_in[1];
  const int*   cu   = (const int*)d_in[2];
  // d_in[3] (max_seqlen) lives on device; grid upper-bounded by total tokens.
  float* out = (float*)d_out;

  const int total = in_sizes[0] / (16 * 64);   // H*D = 1024
  const int B     = in_sizes[2] - 1;
  const int tiles = (total + 15) / 16;         // >= ceil(max_seqlen/16)

  dim3 grid(tiles, 16 /*H*/, B);
  dim3 block(32);
  hipLaunchKernelGGL(fa_varlen_wmma, grid, block, 0, stream, q_p, kv_p, cu, out);
}
